// TDMF_FWI_35519379538393
// MI455X (gfx1250) — compile-verified, matrix operand order checked by reading, not run on hardware
//
#include <hip/hip_runtime.h>
#include <math.h>

// ---------------- problem constants (from reference) ----------------
constexpr int B_    = 32;
constexpr int H_    = 105;
constexpr int W_    = 70;
constexpr int C_    = 128;
constexpr int HW_   = 15;    // H / 7
constexpr int WW_   = 10;    // W / 7
constexpr int NWIN_ = 150;   // HW*WW
constexpr int PQ_   = 49;    // 7*7
constexpr int NH_   = 4;
constexpr int HEAD_ = 32;
constexpr int NQKV_ = 384;   // 3*C
constexpr int SA_   = 136;   // bf16 LDS row stride (multiple of 8 -> 16B aligned)

typedef __attribute__((ext_vector_type(2)))  float  v2f;
typedef __attribute__((ext_vector_type(8)))  float  v8f;
typedef __attribute__((ext_vector_type(4)))  __bf16 v4bf;
typedef __attribute__((ext_vector_type(8)))  __bf16 v8bf;
typedef __attribute__((ext_vector_type(16))) __bf16 v16bf;

// fp32 WMMA: D(16x16) = A(16x4) * B(4x16) + C
__device__ __forceinline__ v8f wmma4(v2f a, v2f b, v8f c) {
  return __builtin_amdgcn_wmma_f32_16x16x4_f32(false, a, false, b, (short)0, c,
                                               false, false);
}
// bf16 XDL WMMA: D(16x16,f32) = A(16x32,bf16) * B(32x16,bf16) + C
__device__ __forceinline__ v8f wmma_bf(v16bf a, v16bf b, v8f c) {
  return __builtin_amdgcn_wmma_f32_16x16x32_bf16(false, a, false, b, (short)0,
                                                 c, false, false);
}
// 16-bit fragment: two contiguous 8-elem K runs at p and p+16 (ISA layout)
__device__ __forceinline__ v16bf frag16(const __bf16* p) {
  v8bf a = *reinterpret_cast<const v8bf*>(p);
  v8bf b = *reinterpret_cast<const v8bf*>(p + 16);
  return __builtin_shufflevector(a, b, 0, 1, 2, 3, 4, 5, 6, 7, 8, 9, 10, 11,
                                 12, 13, 14, 15);
}
__device__ __forceinline__ __bf16 bhi(float f) { return (__bf16)f; }
__device__ __forceinline__ __bf16 blo(float f, __bf16 h) {
  return (__bf16)(f - (float)h);
}

// ============================================================================
// Kernel 0: split w_qkv / w_out into bf16 hi/lo pairs (L2-resident operands).
// ============================================================================
__global__ __launch_bounds__(256) void split_kernel(
    const float* __restrict__ w_qkv, const float* __restrict__ w_out,
    __bf16* __restrict__ whi, __bf16* __restrict__ wlo,
    __bf16* __restrict__ wohi, __bf16* __restrict__ wolo) {
  const int i = blockIdx.x * blockDim.x + threadIdx.x;
  if (i < NQKV_ * C_) {
    const float f = w_qkv[i];
    const __bf16 h = bhi(f);
    whi[i] = h;
    wlo[i] = blo(f, h);
  }
  if (i < C_ * C_) {
    const float f = w_out[i];
    const __bf16 h = bhi(f);
    wohi[i] = h;
    wolo[i] = blo(f, h);
  }
}

// ============================================================================
// Kernel 1: fused window-gather (roll -3,-3) + bf16-split QKV projection.
// One workgroup per (batch, window). Output qkv_ws[b][win][3][h][49][32] fp32.
// ============================================================================
__global__ __launch_bounds__(256) void qkv_kernel(
    const float* __restrict__ x, const __bf16* __restrict__ whi,
    const __bf16* __restrict__ wlo, const float* __restrict__ b_qkv,
    float* __restrict__ qkv_ws) {
  __shared__ __align__(16) __bf16 Ahi[64 * SA_];
  __shared__ __align__(16) __bf16 Alo[64 * SA_];

  const int blk = blockIdx.x;
  const int b   = blk / NWIN_;
  const int win = blk % NWIN_;
  const int wr  = win / WW_;
  const int wc  = win % WW_;
  const int tid = threadIdx.x;

  // ---- gather shifted window, convert to bf16 hi/lo in LDS (pad to 64 rows)
  for (int e = tid; e < 64 * 32; e += 256) {
    const int row = e >> 5;     // token (0..63)
    const int seg = e & 31;     // float4 segment of 128 channels
    float4 val = make_float4(0.f, 0.f, 0.f, 0.f);
    if (row < PQ_) {
      const int pr = row / 7, pc = row % 7;
      const int gr = (wr * 7 + pr + 3) % H_;   // roll(-3) on H
      const int gc = (wc * 7 + pc + 3) % W_;   // roll(-3) on W
      val = reinterpret_cast<const float4*>(
          x + (((size_t)b * H_ + gr) * W_ + gc) * C_)[seg];
    }
    const __bf16 h0 = bhi(val.x), h1 = bhi(val.y), h2 = bhi(val.z),
                 h3 = bhi(val.w);
    v4bf hv = {h0, h1, h2, h3};
    v4bf lv = {blo(val.x, h0), blo(val.y, h1), blo(val.z, h2), blo(val.w, h3)};
    *reinterpret_cast<v4bf*>(&Ahi[row * SA_ + seg * 4]) = hv;
    *reinterpret_cast<v4bf*>(&Alo[row * SA_ + seg * 4]) = lv;
  }
  __syncthreads();

  const int wave = tid >> 5;
  const int lane = tid & 31;
  const int half = lane >> 4;
  const int ln   = lane & 15;
  const int g    = half * 8;    // K sub-group per 16-bit A/B striping

  // 64x384 output = 4 x 24 tiles of 16x16; 8 waves round-robin.
  for (int t = wave; t < 4 * 24; t += 8) {
    const int tm = t / 24, tn = t % 24;
    v8f acc = {};
    const __bf16* ah = Ahi + (size_t)(tm * 16 + ln) * SA_ + g;
    const __bf16* al = Alo + (size_t)(tm * 16 + ln) * SA_ + g;
    const __bf16* bh = whi + (size_t)(tn * 16 + ln) * C_ + g;
    const __bf16* bl = wlo + (size_t)(tn * 16 + ln) * C_ + g;
    for (int kk = 0; kk < C_; kk += 32) {
      const v16bf ahf = frag16(ah + kk);
      const v16bf alf = frag16(al + kk);
      const v16bf bhf = frag16(bh + kk);
      const v16bf blf = frag16(bl + kk);
      acc = wmma_bf(ahf, bhf, acc);   // hi*hi
      acc = wmma_bf(ahf, blf, acc);   // hi*lo
      acc = wmma_bf(alf, bhf, acc);   // lo*hi
    }
    const int n     = tn * 16 + ln;          // 0..383
    const int which = n >> 7;                // 0=q 1=k 2=v
    const int h     = (n >> 5) & 3;
    const int d     = n & 31;
    const float bias = b_qkv[n];
    const size_t base =
        (((size_t)(b * NWIN_ + win) * 3 + which) * NH_ + h) * PQ_ * HEAD_ + d;
    for (int r = 0; r < 8; ++r) {
      const int m = tm * 16 + r + 8 * half;  // C/D layout: rows r / r+8
      if (m < PQ_) qkv_ws[base + (size_t)m * HEAD_] = acc[r] + bias;
    }
  }
}

// ============================================================================
// Kernel 2: fused attention (fp32 WMMA) + bf16-split out projection +
// reshape/roll(+3,+3) scatter. One workgroup per (b, win); one wave per head.
// ============================================================================
__global__ __launch_bounds__(128) void attn_kernel(
    const float* __restrict__ qkv_ws, const float* __restrict__ rel_pos,
    const __bf16* __restrict__ wohi, const __bf16* __restrict__ wolo,
    const float* __restrict__ b_out, float* __restrict__ out) {
  // sim: 4 heads x [49][52] fp32 (40.8 KB); reused later as bf16 attn hi/lo.
  __shared__ __align__(16) float smem[4 * 49 * 52];

  const int blk = blockIdx.x;
  const int b   = blk / NWIN_;
  const int win = blk % NWIN_;
  const int wr  = win / WW_;
  const int wc  = win % WW_;
  const int tid = threadIdx.x;
  const int hd  = tid >> 5;     // head == wave
  const int lane = tid & 31;
  const int half = lane >> 4;
  const int ln   = lane & 15;
  const int g    = half * 8;

  const size_t wbase = (size_t)(b * NWIN_ + win) * 3 * NH_ * PQ_ * HEAD_;
  const float* qptr = qkv_ws + wbase + (size_t)(0 * NH_ + hd) * PQ_ * HEAD_;
  const float* kptr = qkv_ws + wbase + (size_t)(1 * NH_ + hd) * PQ_ * HEAD_;
  const float* vptr = qkv_ws + wbase + (size_t)(2 * NH_ + hd) * PQ_ * HEAD_;
  float* sim = smem + hd * 49 * 52;

  const float scale = 0.17677669529663687f;  // 32^-0.5

  // ---- sim = scale * q @ k^T + rel + prior, shift mask -> -inf -----------
  for (int t = 0; t < 16; ++t) {            // 4 x 4 tiles over 49x49 (pad 64)
    const int tm = t >> 2, tn = t & 3;
    v8f acc = {};
    for (int kk = 0; kk < HEAD_; kk += 4) {
      const int k0 = kk + 2 * half;
      const int m  = tm * 16 + ln;
      const int q  = tn * 16 + ln;
      v2f a = {}, bf = {};
      if (m < PQ_) {
        const float2 t2 = *reinterpret_cast<const float2*>(qptr + m * HEAD_ + k0);
        a.x = t2.x; a.y = t2.y;
      }
      if (q < PQ_) {
        const float2 t2 = *reinterpret_cast<const float2*>(kptr + q * HEAD_ + k0);
        bf.x = t2.x; bf.y = t2.y;
      }
      acc = wmma4(a, bf, acc);
    }
    const int q = tn * 16 + ln;
    if (q < PQ_) {
      const int qr = q / 7, qc = q % 7;
      for (int r = 0; r < 8; ++r) {
        const int p = tm * 16 + r + 8 * half;
        if (p < PQ_) {
          const int pr = p / 7, pc = p % 7;
          const int dr = pr - qr, dc = pc - qc;
          const float d2 = (float)(dr * dr + dc * dc);
          float val = acc[r] * scale +
                      rel_pos[hd * 169 + (dr + 6) * 13 + (dc + 6)] +
                      __expf(-d2 * (9.0f / 98.0f));   // sigma = 7/3 prior
          const bool mask = (wr == HW_ - 1 && ((pr < 4) != (qr < 4))) ||
                            (wc == WW_ - 1 && ((pc < 4) != (qc < 4)));
          sim[p * 52 + q] = mask ? -INFINITY : val;
        }
      }
    }
  }
  __syncthreads();

  // ---- row softmax (each wave owns its head's 49 rows) -------------------
  for (int p = lane; p < PQ_; p += 32) {
    float mx = -INFINITY;
    for (int q = 0; q < PQ_; ++q) mx = fmaxf(mx, sim[p * 52 + q]);
    float s = 0.f;
    for (int q = 0; q < PQ_; ++q) {
      const float e = __expf(sim[p * 52 + q] - mx);
      sim[p * 52 + q] = e;
      s += e;
    }
    const float inv = 1.0f / s;
    for (int q = 0; q < PQ_; ++q) sim[p * 52 + q] *= inv;
  }
  __syncthreads();

  // ---- probs @ v, 4x2 tiles kept in registers (fp32 WMMA, guarded K) -----
  v8f oacc[8];
  {
    v8f z = {};
    for (int i = 0; i < 8; ++i) oacc[i] = z;
  }
  for (int kk = 0; kk < PQ_; kk += 4) {
    const int k0 = kk + 2 * half;
    v2f bfr[2];
    for (int tn = 0; tn < 2; ++tn) {
      const int n = tn * 16 + ln;   // n < 32 always valid
      v2f bf = {};
      if (k0 < PQ_)     bf.x = vptr[k0 * HEAD_ + n];
      if (k0 + 1 < PQ_) bf.y = vptr[(k0 + 1) * HEAD_ + n];
      bfr[tn] = bf;
    }
    for (int tm = 0; tm < 4; ++tm) {
      const int m = tm * 16 + ln;
      v2f a = {};
      if (m < PQ_) {
        if (k0 < PQ_)     a.x = sim[m * 52 + k0];
        if (k0 + 1 < PQ_) a.y = sim[m * 52 + k0 + 1];
      }
      oacc[tm * 2 + 0] = wmma4(a, bfr[0], oacc[tm * 2 + 0]);
      oacc[tm * 2 + 1] = wmma4(a, bfr[1], oacc[tm * 2 + 1]);
    }
  }
  __syncthreads();   // sim dead -> reuse LDS as bf16 attn hi/lo [49][136]

  __bf16* attnhi = reinterpret_cast<__bf16*>(smem);
  __bf16* attnlo = attnhi + PQ_ * SA_;          // 13328 B offset (16B aligned)
  for (int tm = 0; tm < 4; ++tm)
    for (int tn = 0; tn < 2; ++tn) {
      const v8f acc = oacc[tm * 2 + tn];
      const int n = tn * 16 + ln;
      for (int r = 0; r < 8; ++r) {
        const int m = tm * 16 + r + 8 * half;
        if (m < PQ_) {
          const float f = acc[r];
          const __bf16 h = bhi(f);
          attnhi[m * SA_ + hd * 32 + n] = h;
          attnlo[m * SA_ + hd * 32 + n] = blo(f, h);
        }
      }
    }
  __syncthreads();

  // ---- out projection (49x128 @ 128x128, bf16 split) + fused scatter -----
  for (int t = hd; t < 32; t += 4) {        // 4 m-tiles x 8 n-tiles
    const int tm = t >> 3, tn = t & 7;
    const int m  = tm * 16 + ln;
    v8f acc = {};
    const __bf16* bh = wohi + (size_t)(tn * 16 + ln) * C_ + g;
    const __bf16* bl = wolo + (size_t)(tn * 16 + ln) * C_ + g;
    for (int kk = 0; kk < C_; kk += 32) {
      v16bf ahf = {}, alf = {};
      if (m < PQ_) {
        ahf = frag16(attnhi + m * SA_ + g + kk);
        alf = frag16(attnlo + m * SA_ + g + kk);
      }
      const v16bf bhf = frag16(bh + kk);
      const v16bf blf = frag16(bl + kk);
      acc = wmma_bf(ahf, bhf, acc);
      acc = wmma_bf(ahf, blf, acc);
      acc = wmma_bf(alf, bhf, acc);
    }
    const int n = tn * 16 + ln;
    const float bias = b_out[n];
    for (int r = 0; r < 8; ++r) {
      const int mm = tm * 16 + r + 8 * half;
      if (mm < PQ_) {
        const int pr = mm / 7, pc = mm % 7;
        // out[b, WS, WW*WS, RATIO*C] after transpose(0,2,4,3,5,1,6)+roll(+3,+3)
        const int h2 = (pr + 3) % 7;
        const int w2 = (wc * 7 + pc + 3) % 70;
        const int cc = wr * C_ + n;
        out[(((size_t)b * 7 + h2) * 70 + w2) * (15 * C_) + cc] = acc[r] + bias;
      }
    }
  }
}

// ============================================================================
extern "C" void kernel_launch(void* const* d_in, const int* in_sizes, int n_in,
                              void* d_out, int out_size, void* d_ws,
                              size_t ws_size, hipStream_t stream) {
  (void)in_sizes; (void)n_in; (void)out_size; (void)ws_size;
  const float* x       = (const float*)d_in[0];
  const float* w_qkv   = (const float*)d_in[1];
  const float* b_qkv   = (const float*)d_in[2];
  const float* rel_pos = (const float*)d_in[3];
  const float* w_out   = (const float*)d_in[4];
  const float* b_out   = (const float*)d_in[5];

  // workspace layout: fp32 q/k/v staging, then bf16 hi/lo weight splits
  const size_t qkv_elems = (size_t)B_ * NWIN_ * 3 * NH_ * PQ_ * HEAD_;
  float*  qkv_ws = (float*)d_ws;
  __bf16* whi  = (__bf16*)(qkv_ws + qkv_elems);
  __bf16* wlo  = whi + (size_t)NQKV_ * C_;
  __bf16* wohi = wlo + (size_t)NQKV_ * C_;
  __bf16* wolo = wohi + (size_t)C_ * C_;

  split_kernel<<<dim3((NQKV_ * C_ + 255) / 256), dim3(256), 0, stream>>>(
      w_qkv, w_out, whi, wlo, wohi, wolo);
  qkv_kernel<<<dim3(B_ * NWIN_), dim3(256), 0, stream>>>(x, whi, wlo, b_qkv,
                                                         qkv_ws);
  attn_kernel<<<dim3(B_ * NWIN_), dim3(128), 0, stream>>>(
      qkv_ws, rel_pos, wohi, wolo, b_out, (float*)d_out);
}
